// FasterRCNN_87703232184711
// MI455X (gfx1250) — matrix operand order repaired and checked by reading, not compiled, not run on hardware
//
#include <hip/hip_runtime.h>
#include <hip/hip_bf16.h>

// ---------------- CDNA5 WMMA types ----------------
typedef __attribute__((ext_vector_type(16))) __bf16         v16bf;
typedef __attribute__((ext_vector_type(8)))  float          v8f;
typedef __attribute__((ext_vector_type(8)))  unsigned int   v8u;

static_assert(sizeof(v16bf) == 32, "v16bf size");
static_assert(sizeof(v8f)   == 32, "v8f size");

// Problem constants (match reference)
#define N_B     4
#define N_R     256
#define N_ROI   3072      // B * 3 * R
#define N_CH    256
#define N_POS   49        // 7x7
#define K_CONV  2304      // 256 * 9
#define KC_CONV 72        // 2304 / 32
#define K_FC    12544     // 256 * 49
#define KC_FC   392       // 12544 / 32
#define N_CLS   21        // NUM_CLASSES + 1

__device__ __forceinline__ unsigned short f32_to_bf16(float f) {
  unsigned int u = __builtin_bit_cast(unsigned int, f);
  unsigned int r = u + 0x7FFFu + ((u >> 16) & 1u);   // round-to-nearest-even
  return (unsigned short)(r >> 16);
}

__device__ __forceinline__ v16bf load_frag_contig(const unsigned short* p) {
  // 16 contiguous bf16 per lane = 2 x b128 (global_load_b128 or ds_load_b128)
  const uint4* q = (const uint4*)p;
  uint4 lo = q[0];
  uint4 hi = q[1];
  v8u u;
  u[0] = lo.x; u[1] = lo.y; u[2] = lo.z; u[3] = lo.w;
  u[4] = hi.x; u[5] = hi.y; u[6] = hi.z; u[7] = hi.w;
  return __builtin_bit_cast(v16bf, u);
}

__device__ __forceinline__ v16bf load_frag_split(const unsigned short* p0,
                                                 const unsigned short* p1) {
  // A fragment: 8 halves at p0 (K = base..base+7), 8 at p1 (K = base+8..+15)
  uint4 lo = *(const uint4*)p0;
  uint4 hi = *(const uint4*)p1;
  v8u u;
  u[0] = lo.x; u[1] = lo.y; u[2] = lo.z; u[3] = lo.w;
  u[4] = hi.x; u[5] = hi.y; u[6] = hi.z; u[7] = hi.w;
  return __builtin_bit_cast(v16bf, u);
}

// ---- gfx1250 async global->LDS (ASYNCcnt) via inline asm ----
// GVS addressing: mem = SADDR(64) + VADDR(32-bit byte offset); VDST = LDS byte addr
__device__ __forceinline__ void async_ld_b128(unsigned lds_addr, unsigned gbyte_off,
                                              const unsigned short* sbase) {
  asm volatile("global_load_async_to_lds_b128 %0, %1, %2"
               :
               : "v"(lds_addr), "v"(gbyte_off), "s"((unsigned long long)(size_t)sbase)
               : "memory");
}

__device__ __forceinline__ void wait_async0() {
  asm volatile("s_wait_asynccnt 0x0" ::: "memory");
}

// -------------------------------------------------------------------------
// Prep: swizzle conv weights (Cout=256, Cin=256, 3, 3) -> B-fragment layout.
// GEMM K ordering is TAP-MAJOR: k' = tap*256 + cin, tap = ky*3+kx, so each
// 32-wide K-chunk sits inside one tap (chunk kc: tap = kc>>3, c = (kc&7)*32+koff).
// Layout index: (((ntile*72 + kchunk)*32 + lane)*16 + i), bf16
// B-matrix element mapping (32x16, 16-bit): lane l, half i:
//   K = (l>>4)*16 + i ; N = (l & 15)
// -------------------------------------------------------------------------
__global__ __launch_bounds__(256) void prep_convw_kernel(
    const float* __restrict__ w, unsigned short* __restrict__ out) {
  int e = blockIdx.x * 256 + threadIdx.x;
  if (e >= K_CONV * N_CH) return;
  int i     = e & 15;
  int l     = (e >> 4) & 31;
  int kc    = (e >> 9) % KC_CONV;
  int ntile = e / (512 * KC_CONV);
  int koff  = ((l >> 4) << 4) + i;
  int k     = kc * 32 + koff;                 // tap-major K index
  int n     = ntile * 16 + (l & 15);          // output channel
  int tap   = k >> 8;                         // 0..8
  int c     = k & 255;                        // input channel
  int ky    = tap / 3;
  int kx    = tap - ky * 3;
  float v   = w[((n * N_CH + c) * 3 + ky) * 3 + kx];
  out[e] = f32_to_bf16(v);
}

// -------------------------------------------------------------------------
// Prep: swizzle fc weights (21, 12544) -> B-fragment layout, N padded to 32
// Layout index: (((kchunk*2 + ntile)*32 + lane)*16 + i)
// -------------------------------------------------------------------------
__global__ __launch_bounds__(256) void prep_fcw_kernel(
    const float* __restrict__ w, unsigned short* __restrict__ out) {
  int e = blockIdx.x * 256 + threadIdx.x;
  if (e >= KC_FC * 2 * 32 * 16) return;
  int i     = e & 15;
  int l     = (e >> 4) & 31;
  int ntile = (e >> 9) & 1;
  int kc    = e >> 10;
  int koff  = ((l >> 4) << 4) + i;
  int k     = kc * 32 + koff;
  int n     = ntile * 16 + (l & 15);
  float v   = (n < N_CLS) ? w[n * K_FC + k] : 0.0f;
  out[e] = f32_to_bf16(v);
}

// -------------------------------------------------------------------------
// Fused ROI-Align + conv3x3(+bias,ReLU) as implicit GEMM via WMMA.
// One block (256 threads, 8 waves) per ROI.
//   Phase 1: roi-align into LDS tile sFeat[pos][chan] (bf16); meanwhile the
//            first B weight panel streams in via global_load_async_to_lds.
//   Phase 2: im2col GEMM, tap-major K: per chunk the A fragment is 16
//            CONTIGUOUS channels -> 2 ds_load_b128 (zero-pad via LDS zero
//            slab + one address cndmask). B panels double-buffered (ASYNCcnt).
// Output: convout[roi][n*49 + m] bf16 (= flatten(C,7,7) row per ROI)
// -------------------------------------------------------------------------
__global__ __launch_bounds__(256) void roi_conv_kernel(
    const float* __restrict__ f3, const float* __restrict__ f4,
    const float* __restrict__ f5,
    const float* __restrict__ p3, const float* __restrict__ p4,
    const float* __restrict__ p5,
    const float* __restrict__ conv_b,
    const unsigned short* __restrict__ Wsw,
    unsigned short* __restrict__ convout) {
  __shared__ __align__(16) unsigned short sFeat[N_POS * N_CH];  // [pos][chan] 25088B
  __shared__ __align__(16) unsigned short sZero[64];            // zero slab (128B)
  __shared__ __align__(16) unsigned short sB[2][16 * 32 * 16];  // 2 x 16KB B panels
  __shared__ int   sP0[28];   // [0..13] y, [14..27] x
  __shared__ int   sP1[28];
  __shared__ float sL[28];
  __shared__ int   sV[28];

  const int r   = blockIdx.x;
  const int b   = r / 768;
  const int lvl = (r / 256) % 3;
  const int ri  = r % 256;
  const int tid = threadIdx.x;

  // ---- cooperative async stage of one 16KB B panel (all 256 threads) ----
  // panel layout in LDS mirrors Wsw chunk: [ntile(16)][lane(32)][16 halves]
  const int stg_nt  = tid >> 4;   // 0..15
  const int stg_sub = tid & 15;   // 0..15 (64B each)
  auto stage_B = [&](unsigned short* dst, int kc) {
    unsigned gb  = (unsigned)((stg_nt * KC_CONV + kc) * 1024 + stg_sub * 64);
    unsigned la  = (unsigned)(size_t)dst + (unsigned)(stg_nt * 1024 + stg_sub * 64);
#pragma unroll
    for (int q = 0; q < 4; ++q)
      async_ld_b128(la + q * 16, gb + q * 16, Wsw);
  };
  stage_B(&sB[0][0], 0);   // overlaps with ROI-align below

  const float* feat;
  const float* props;
  int H;
  float scale;
  if (lvl == 0)      { feat = f3; props = p3; H = 64; scale = 0.125f;   }
  else if (lvl == 1) { feat = f4; props = p4; H = 32; scale = 0.0625f;  }
  else               { feat = f5; props = p5; H = 16; scale = 0.03125f; }
  const int W = H;
  const float* box = props + (size_t)(b * N_R + ri) * 4;

  // ---- sample-grid parameters (aligned=True roi_align, G=14) ----
  if (tid < 28) {
    int axis = tid / 14;           // 0 = y, 1 = x
    int g    = tid % 14;
    float c1 = box[axis == 0 ? 1 : 0] * scale - 0.5f;
    float c2 = box[axis == 0 ? 3 : 2] * scale - 0.5f;
    float pos = c1 + (g + 0.5f) * (1.0f / 14.0f) * (c2 - c1);
    int valid = (pos >= -1.0f) && (pos <= (float)H);
    float pc  = fminf(fmaxf(pos, 0.0f), (float)(H - 1));
    float p0f = floorf(pc);
    int p0 = (int)p0f;
    int p1 = p0 + 1 < H - 1 ? p0 + 1 : H - 1;
    sP0[tid] = p0;
    sP1[tid] = p1;
    sL[tid]  = pc - p0f;
    sV[tid]  = valid;
  }
  if (tid < 32) ((unsigned int*)sZero)[tid] = 0u;
  __syncthreads();

  // ---- phase 1: roi align; thread == channel; tile stored [pos][chan] ----
  {
    const float* fp = feat + (size_t)(b * N_CH + tid) * H * W;
    for (int bin = 0; bin < N_POS; ++bin) {
      int py = bin / 7, px = bin % 7;
      float acc = 0.0f;
#pragma unroll
      for (int a = 0; a < 2; ++a) {
        int gy = py * 2 + a;
        int y0 = sP0[gy], y1 = sP1[gy];
        float ly = sL[gy];
        int vy = sV[gy];
#pragma unroll
        for (int bx = 0; bx < 2; ++bx) {
          int gx = 14 + px * 2 + bx;
          if (vy && sV[gx]) {
            int x0 = sP0[gx], x1 = sP1[gx];
            float lx = sL[gx];
            float hy = 1.0f - ly, hx = 1.0f - lx;
            acc += fp[y0 * W + x0] * hy * hx + fp[y0 * W + x1] * hy * lx +
                   fp[y1 * W + x0] * ly * hx + fp[y1 * W + x1] * ly * lx;
          }
        }
      }
      sFeat[bin * N_CH + tid] = f32_to_bf16(acc * 0.25f);
    }
  }
  wait_async0();        // first B panel landed
  __syncthreads();      // publish sFeat + sB[0] to all waves

  // ---- phase 2: implicit-GEMM conv via WMMA (tap-major K) ----
  const int wave    = tid >> 5;
  const int lane    = tid & 31;
  const int mtile   = wave & 3;            // 4 M-tiles of 16 (rows 0..63)
  const int ntbase  = (wave >> 2) * 8;     // 2 groups x 8 N-tiles
  const int mrow    = mtile * 16 + (lane & 15);
  const int halfsel = lane >> 4;
  const int py = mrow / 7;
  const int px = mrow - py * 7;
  const bool mvalid = (mrow < N_POS);

  v8f acc[8];
#pragma unroll
  for (int t = 0; t < 8; ++t)
#pragma unroll
    for (int j = 0; j < 8; ++j) acc[t][j] = 0.0f;

  for (int kc = 0; kc < KC_CONV; ++kc) {
    const unsigned short* curB = &sB[kc & 1][0];
    if (kc + 1 < KC_CONV) stage_B(&sB[(kc + 1) & 1][0], kc + 1);  // async DMA next panel

    // A fragment: chunk kc covers tap = kc>>3, channels cbase..cbase+31.
    // Per lane the 16 A halves are contiguous channels -> 2 x ds_load_b128.
    int tap   = kc >> 3;
    int cbase = (kc & 7) * 32;
    int ky = tap / 3;
    int kx = tap - ky * 3;
    int y  = py + ky - 1;
    int x  = px + kx - 1;
    bool ok = mvalid && ((unsigned)y < 7u) && ((unsigned)x < 7u);
    const unsigned short* abase =
        ok ? (sFeat + (y * 7 + x) * N_CH + cbase + halfsel * 16) : (sZero + halfsel * 16);
    v16bf afrag = load_frag_split(abase, abase + 8);

    // 8 N-tiles in two groups of 4: preload B frags (ds_load_b128 clause),
    // then back-to-back WMMAs (no D->A/B RAW between them)
#pragma unroll
    for (int g = 0; g < 2; ++g) {
      v16bf bfr[4];
#pragma unroll
      for (int j = 0; j < 4; ++j)
        bfr[j] = load_frag_contig(curB + (((ntbase + g * 4 + j) * 32 + lane) << 4));
#pragma unroll
      for (int j = 0; j < 4; ++j)
        acc[g * 4 + j] = __builtin_amdgcn_wmma_f32_16x16x32_bf16(
            false, afrag, false, bfr[j], (short)0, acc[g * 4 + j], false, false);
    }

    wait_async0();      // next panel complete (this wave's DMAs)
    __syncthreads();    // all waves done reading curB / DMA visible block-wide
  }

  // ---- epilogue: bias + relu, store bf16 row [roi][n*49 + m] ----
#pragma unroll
  for (int nt = 0; nt < 8; ++nt) {
    int n = (ntbase + nt) * 16 + (lane & 15);
    float bias = conv_b[n];
    unsigned short* orow = convout + (size_t)r * K_FC + n * N_POS;
#pragma unroll
    for (int j = 0; j < 8; ++j) {
      int m = mtile * 16 + halfsel * 8 + j;     // C/D layout: M = vgpr + half*8
      if (m < N_POS) {
        float v = acc[nt][j] + bias;
        orow[m] = f32_to_bf16(v > 0.0f ? v : 0.0f);
      }
    }
  }
}

// -------------------------------------------------------------------------
// FC head: logits = convout(3072 x 12544) @ fcW^T (+bias), N padded 21->32
// 48 blocks x 8 waves; each wave owns one 16x16 tile (4 M-tiles x 2 N-tiles)
// -------------------------------------------------------------------------
__global__ __launch_bounds__(256) void fc_kernel(
    const unsigned short* __restrict__ convout,
    const unsigned short* __restrict__ fcWsw,
    const float* __restrict__ fc_b,
    float* __restrict__ logits) {
  const int tid   = threadIdx.x;
  const int wave  = tid >> 5;
  const int lane  = tid & 31;
  const int mtile = wave >> 1;
  const int ntile = wave & 1;
  const int mbase = blockIdx.x * 64 + mtile * 16;
  const int m     = mbase + (lane & 15);
  const int halfsel = lane >> 4;

  v8f acc;
#pragma unroll
  for (int j = 0; j < 8; ++j) acc[j] = 0.0f;

  const unsigned short* arow = convout + (size_t)m * K_FC;
  for (int kc = 0; kc < KC_FC; ++kc) {
    __builtin_prefetch(arow + (kc + 1) * 32, 0, 1);  // global_prefetch_b8
    const unsigned short* a0 = arow + kc * 32 + halfsel * 16;      // K base..+7
    v16bf afrag = load_frag_split(a0, a0 + 8);                     // K +8..+15
    v16bf bfrag = load_frag_contig(
        fcWsw + (((size_t)kc * 2 + ntile) * 32 + lane) * 16);
    acc = __builtin_amdgcn_wmma_f32_16x16x32_bf16(
        false, afrag, false, bfrag, (short)0, acc, false, false);
  }

  int n = ntile * 16 + (lane & 15);
  if (n < N_CLS) {
    float bias = fc_b[n];
#pragma unroll
    for (int j = 0; j < 8; ++j) {
      int row = mbase + halfsel * 8 + j;
      logits[(size_t)row * N_CLS + n] = acc[j] + bias;
    }
  }
}

// -------------------------------------------------------------------------
// Proposal <-> GT IoU matching (2nd stage): one thread per proposal
// -------------------------------------------------------------------------
__global__ __launch_bounds__(256) void match_kernel(
    const float* __restrict__ p3, const float* __restrict__ p4,
    const float* __restrict__ p5, const float* __restrict__ gt,
    float* __restrict__ outm) {
  int t = blockIdx.x * 256 + threadIdx.x;
  if (t >= N_B * 768) return;
  int b   = t / 768;
  int p   = t % 768;
  int lvl = p / 256;
  int i   = p % 256;
  const float* props = (lvl == 0) ? p3 : ((lvl == 1) ? p4 : p5);
  const float* box = props + (size_t)(b * N_R + i) * 4;
  float x1 = box[0], y1 = box[1], x2 = box[2], y2 = box[3];
  float a1 = (x2 - x1) * (y2 - y1);

  float best = -1.0f;
  int bestj = 0;
  for (int j = 0; j < 40; ++j) {
    const float* g = gt + (size_t)(b * 40 + j) * 5;
    float iou = -1.0f;
    if (g[4] != -1.0f) {
      float tx1 = fmaxf(x1, g[0]), ty1 = fmaxf(y1, g[1]);
      float tx2 = fminf(x2, g[2]), ty2 = fminf(y2, g[3]);
      float iw = fmaxf(tx2 - tx1, 0.0f);
      float ih = fmaxf(ty2 - ty1, 0.0f);
      float inter = iw * ih;
      float a2 = (g[2] - g[0]) * (g[3] - g[1]);
      iou = inter / (a1 + a2 - inter);
    }
    if (iou > best) { best = iou; bestj = j; }
  }
  const float* g = gt + (size_t)(b * 40 + bestj) * 5;
  float o[5];
#pragma unroll
  for (int k = 0; k < 5; ++k) o[k] = g[k];
  if (best <= 0.4f) {
#pragma unroll
    for (int k = 0; k < 5; ++k) o[k] = -1.0f;
  } else if (best < 0.6f) {
#pragma unroll
    for (int k = 0; k < 5; ++k) o[k] = -1e8f;
  }
#pragma unroll
  for (int k = 0; k < 5; ++k) outm[(size_t)t * 5 + k] = o[k];
}

// -------------------------------------------------------------------------
extern "C" void kernel_launch(void* const* d_in, const int* in_sizes, int n_in,
                              void* d_out, int out_size, void* d_ws, size_t ws_size,
                              hipStream_t stream) {
  const float* f3     = (const float*)d_in[0];
  const float* f4     = (const float*)d_in[1];
  const float* f5     = (const float*)d_in[2];
  const float* p3     = (const float*)d_in[3];
  const float* p4     = (const float*)d_in[4];
  const float* p5     = (const float*)d_in[5];
  const float* gt     = (const float*)d_in[6];
  const float* conv_w = (const float*)d_in[7];
  const float* conv_b = (const float*)d_in[8];
  const float* fc_w   = (const float*)d_in[9];
  const float* fc_b   = (const float*)d_in[10];
  float* out = (float*)d_out;

  // workspace layout (all 256B-aligned)
  char* ws = (char*)d_ws;
  const size_t convout_bytes = (size_t)N_ROI * K_FC * sizeof(unsigned short); // 77,070,336
  const size_t convw_bytes   = (size_t)K_CONV * N_CH * sizeof(unsigned short); // 1,179,648
  unsigned short* convout = (unsigned short*)ws;
  unsigned short* Wsw     = (unsigned short*)(ws + convout_bytes);
  unsigned short* fcWsw   = (unsigned short*)(ws + convout_bytes + convw_bytes);

  // 1) swizzle weights into WMMA B-fragment layouts (tap-major K for conv)
  prep_convw_kernel<<<(K_CONV * N_CH) / 256, 256, 0, stream>>>(conv_w, Wsw);
  prep_fcw_kernel<<<(KC_FC * 2 * 32 * 16) / 256, 256, 0, stream>>>(fc_w, fcWsw);

  // 2) fused roi-align + conv3x3 + bias + relu (bf16 WMMA, async LDS staging)
  roi_conv_kernel<<<N_ROI, 256, 0, stream>>>(f3, f4, f5, p3, p4, p5,
                                             conv_b, Wsw, convout);

  // 3) FC classification head (bf16 WMMA GEMM) -> logits (3072 x 21)
  fc_kernel<<<N_ROI / 64, 256, 0, stream>>>(convout, fcWsw, fc_b, out);

  // 4) proposal/GT matching -> matched (3072 x 5) after logits
  match_kernel<<<(N_B * 768 + 255) / 256, 256, 0, stream>>>(
      p3, p4, p5, gt, out + (size_t)N_ROI * N_CLS);
}
